// PPFDecoder_75857712382173
// MI455X (gfx1250) — matrix operand order repaired and checked by reading, not compiled, not run on hardware
//
#include <hip/hip_runtime.h>
#include <hip/hip_bf16.h>

// ---------------------------------------------------------------------------
// PPF decoder for MI455X (gfx1250, wave32).
// Pipeline:
//   1) topk          : 40-NN per point from dist row (argmin w/ index tiebreak)
//   2) kernmlp  x2   : per-(i,k) kernel MLP 3->32->64->32->32->32 (VALU, wave/pair)
//   3) nf0 / gather  : neighbor features (norm,cos) / gathered feat
//   4) conv_out x2   : agg einsum + oW linear + relu + aggr g
//   5) gmax/feat x2  : global max prop + concat
//   6) pair_mlp      : 768x768 pair residual MLP via v_wmma_f32_16x16x32_f16
// ---------------------------------------------------------------------------

#define NPTS 768
#define KNN  40

typedef _Float16 half16 __attribute__((ext_vector_type(16)));
typedef _Float16 half8  __attribute__((ext_vector_type(8)));
typedef float    float8 __attribute__((ext_vector_type(8)));

__device__ __forceinline__ float8 f8zero() {
  return (float8){0.f, 0.f, 0.f, 0.f, 0.f, 0.f, 0.f, 0.f};
}

__device__ __forceinline__ void lds_fence() {
  __asm__ volatile("s_wait_dscnt 0" ::: "memory");
}

// A fragment (16x32 f16): lane<16 -> M=lane, K=kb..kb+15;
// lane>=16 -> M=lane-16, K=kb+16..kb+31
__device__ __forceinline__ half16 ldA(const _Float16* base, int ldk, int m,
                                      int kbase, bool hi) {
  return *(const half16*)(base + m * ldk + kbase + (hi ? 16 : 0));
}

// B fragment (32x16 f16) from LDS weight stored transposed [N][ldk]:
// lane<16 -> N=lane, K=kb..kb+15 ; lane>=16 -> N=lane-16, K=kb+16..kb+31
__device__ __forceinline__ half16 ldB(const _Float16* wt, int ldk, int n,
                                      int kbase, bool hi) {
  return *(const half16*)(wt + n * ldk + kbase + (hi ? 16 : 0));
}

__device__ __forceinline__ float8 wmma16(half16 a, half16 b, float8 c) {
  return __builtin_amdgcn_wmma_f32_16x16x32_f16(false, a, false, b, (short)0, c,
                                                false, false);
}

// convert W[kin][kout] (row-major f32) -> dst[Ncap][Kpad] f16 transposed, 0-pad
__device__ void cvtW(_Float16* dst, const float* __restrict__ W, int kin,
                     int kout, int Kpad, int Ncap, int tid, int nth) {
  for (int idx = tid; idx < Ncap * Kpad; idx += nth) {
    int n = idx / Kpad, k = idx - n * Kpad;
    float v = (k < kin && n < kout) ? W[k * kout + n] : 0.0f;
    dst[idx] = (_Float16)v;
  }
}

__device__ void cvtB(float* dst, const float* __restrict__ b, int n, int ncap,
                     int tid, int nth) {
  for (int e = tid; e < ncap; e += nth) dst[e] = (e < n) ? b[e] : 0.0f;
}

// ---------------------------------------------------------------------------
// 1) top-40 nearest neighbors (ascending dist, lower index on ties)
// ---------------------------------------------------------------------------
__global__ __launch_bounds__(64) void topk_kernel(const float* __restrict__ dist,
                                                  int* __restrict__ nbrs) {
  __shared__ float v[NPTS];
  __shared__ float cv[64];
  __shared__ int ci[64];
  const int i = blockIdx.x, t = threadIdx.x;
  for (int j = t; j < NPTS; j += 64) v[j] = dist[i * NPTS + j];
  __syncthreads();
  for (int it = 0; it < KNN; ++it) {
    float best = 3.4e38f;
    int bi = 1 << 30;
    for (int j = t; j < NPTS; j += 64) {
      float x = v[j];
      if (x < best || (x == best && j < bi)) { best = x; bi = j; }
    }
    cv[t] = best; ci[t] = bi;
    __syncthreads();
    if (t == 0) {
      float b = cv[0]; int bj = ci[0];
      for (int u = 1; u < 64; ++u)
        if (cv[u] < b || (cv[u] == b && ci[u] < bj)) { b = cv[u]; bj = ci[u]; }
      nbrs[i * KNN + it] = bj;
      v[bj] = 3.4e38f;
    }
    __syncthreads();
  }
}

// ---------------------------------------------------------------------------
// 2) kernel MLP (3->32->64->32->32->32), one wave per (i,k) pair
// ---------------------------------------------------------------------------
__global__ __launch_bounds__(256) void kernmlp_kernel(
    const float* __restrict__ pc, const int* __restrict__ nbrs,
    const float* __restrict__ W0, const float* __restrict__ b0,
    const float* __restrict__ W1, const float* __restrict__ b1,
    const float* __restrict__ W2, const float* __restrict__ b2,
    const float* __restrict__ W3, const float* __restrict__ b3,
    const float* __restrict__ W4, const float* __restrict__ b4,
    float* __restrict__ kern) {
  __shared__ float A[8][64];
  __shared__ float Bs[8][64];
  const int wave = threadIdx.x >> 5, lane = threadIdx.x & 31;
  const int pair = blockIdx.x * 8 + wave;  // grid = 768*40/8
  const int i = pair / KNN;
  const int j = nbrs[pair];
  const float rx = pc[j * 3 + 0] - pc[i * 3 + 0];
  const float ry = pc[j * 3 + 1] - pc[i * 3 + 1];
  const float rz = pc[j * 3 + 2] - pc[i * 3 + 2];
  // L0: 3->32
  float h = rx * W0[0 * 32 + lane] + ry * W0[1 * 32 + lane] +
            rz * W0[2 * 32 + lane] + b0[lane];
  A[wave][lane] = fmaxf(h, 0.f);
  __syncthreads();
  // L1: 32->64
  float s0 = b1[lane], s1 = b1[lane + 32];
  for (int c = 0; c < 32; ++c) {
    float a = A[wave][c];
    s0 += a * W1[c * 64 + lane];
    s1 += a * W1[c * 64 + lane + 32];
  }
  Bs[wave][lane] = fmaxf(s0, 0.f);
  Bs[wave][lane + 32] = fmaxf(s1, 0.f);
  __syncthreads();
  // L2: 64->32
  float s = b2[lane];
  for (int c = 0; c < 64; ++c) s += Bs[wave][c] * W2[c * 32 + lane];
  A[wave][lane] = fmaxf(s, 0.f);
  __syncthreads();
  // L3: 32->32
  s = b3[lane];
  for (int c = 0; c < 32; ++c) s += A[wave][c] * W3[c * 32 + lane];
  Bs[wave][lane] = fmaxf(s, 0.f);
  __syncthreads();
  // L4: 32->32 (no relu)
  s = b4[lane];
  for (int c = 0; c < 32; ++c) s += Bs[wave][c] * W4[c * 32 + lane];
  kern[pair * 32 + lane] = s;
}

// ---------------------------------------------------------------------------
// 3a) neighbor features for spconv0: [norm, normal_cos]
// ---------------------------------------------------------------------------
__global__ void nf0_kernel(const float* __restrict__ pc,
                           const float* __restrict__ nrm,
                           const int* __restrict__ nbrs,
                           float* __restrict__ nf) {
  int p = blockIdx.x * 256 + threadIdx.x;
  if (p >= NPTS * KNN) return;
  int i = p / KNN;
  int j = nbrs[p];
  float dx = pc[j * 3 + 0] - pc[i * 3 + 0];
  float dy = pc[j * 3 + 1] - pc[i * 3 + 1];
  float dz = pc[j * 3 + 2] - pc[i * 3 + 2];
  nf[p * 2 + 0] = sqrtf(dx * dx + dy * dy + dz * dz);
  nf[p * 2 + 1] = nrm[j * 3 + 0] * nrm[i * 3 + 0] +
                  nrm[j * 3 + 1] * nrm[i * 3 + 1] +
                  nrm[j * 3 + 2] * nrm[i * 3 + 2];
}

// 3b) gather feat rows for spconv1
__global__ void gather_feat_kernel(const float* __restrict__ feat,
                                   const int* __restrict__ nbrs,
                                   float* __restrict__ nf) {
  int idx = blockIdx.x * 256 + threadIdx.x;
  if (idx >= NPTS * KNN * 40) return;
  int c = idx % 40;
  int p = idx / 40;
  nf[idx] = feat[nbrs[p] * 40 + c];
}

// ---------------------------------------------------------------------------
// 4) conv combine: agg = (1/40) sum_k kern[k][r]*nf[k][c]; out=relu(agg@oW+ob)
//    g = out @ aW + ab
// ---------------------------------------------------------------------------
__global__ __launch_bounds__(128) void conv_out_kernel(
    const float* __restrict__ kern, const float* __restrict__ nf, int C,
    const float* __restrict__ oW, const float* __restrict__ ob,
    const float* __restrict__ aW, const float* __restrict__ ab,
    float* __restrict__ convout, float* __restrict__ g) {
  __shared__ float sk[KNN * 32];
  __shared__ float sn[KNN * 40];
  __shared__ float sa[32 * 40];
  __shared__ float so[32];
  const int i = blockIdx.x, t = threadIdx.x;
  for (int e = t; e < KNN * 32; e += 128) sk[e] = kern[i * KNN * 32 + e];
  for (int e = t; e < KNN * C; e += 128) sn[e] = nf[i * KNN * C + e];
  __syncthreads();
  for (int e = t; e < 32 * C; e += 128) {
    int r = e / C, c = e - r * C;
    float s = 0.f;
    for (int k = 0; k < KNN; ++k) s += sk[k * 32 + r] * sn[k * C + c];
    sa[e] = s * (1.0f / (float)KNN);
  }
  __syncthreads();
  if (t < 32) {
    float s = ob[t];
    const int F = 32 * C;
    for (int e = 0; e < F; ++e) s += sa[e] * oW[e * 32 + t];
    s = fmaxf(s, 0.f);
    so[t] = s;
    convout[i * 32 + t] = s;
  }
  __syncthreads();
  if (t < 8) {
    float s = ab[t];
    for (int o = 0; o < 32; ++o) s += so[o] * aW[o * 8 + t];
    g[i * 8 + t] = s;
  }
}

__global__ __launch_bounds__(256) void gmax_kernel(const float* __restrict__ g,
                                                   float* __restrict__ gmax) {
  __shared__ float red[256];
  const int t = threadIdx.x, c = t & 7;
  float m = -3.4e38f;
  for (int i = t >> 3; i < NPTS; i += 32) m = fmaxf(m, g[i * 8 + c]);
  red[t] = m;
  __syncthreads();
  if (t < 8) {
    float mm = red[t];
    for (int u = t + 8; u < 256; u += 8) mm = fmaxf(mm, red[u]);
    gmax[t] = mm;
  }
}

__global__ void feat_kernel(const float* __restrict__ convout,
                            const float* __restrict__ gmax,
                            float* __restrict__ feat) {
  int idx = blockIdx.x * 256 + threadIdx.x;
  if (idx >= NPTS * 40) return;
  int i = idx / 40, c = idx - i * 40;
  feat[idx] = (c < 32) ? convout[i * 32 + c] : gmax[c - 32];
}

// ---------------------------------------------------------------------------
// 6) pair residual MLP via WMMA. Block = point i, 4 waves, each wave owns
//    16-j tiles. All GEMMs are v_wmma_f32_16x16x32_f16.
// ---------------------------------------------------------------------------
#define WPB 4
// LDS weight region offsets (halves), layout [N][Kpad]
#define OW1_0 0       // 32 x 96
#define OW0_0 3072    // 32 x 96
#define OW2_0 6144    // 32 x 32
#define OW1_1 7168    // 32 x 32
#define OW2_1 8192    // 32 x 32
#define OW1_2 9216    // 16 x 32
#define OW0_2 9728    // 16 x 32
#define OW2_2 10240   // 16 x 32 (K>=16 zero)
#define OWF   10752   // 48 x 32 (N>=40 zero, K>=16 zero)
// bias offsets (floats)
#define BB1_0 0
#define BB0_0 32
#define BB2_0 64
#define BB1_1 96
#define BB2_1 128
#define BB1_2 160
#define BB0_2 176
#define BB2_2 192
#define BBF   208

__global__ __launch_bounds__(128) void pair_mlp_kernel(
    const float* __restrict__ pc, const float* __restrict__ nrm,
    const float* __restrict__ dist, const float* __restrict__ feat,
    const float* __restrict__ r0W0, const float* __restrict__ r0W1,
    const float* __restrict__ r0W2, const float* __restrict__ r0b0,
    const float* __restrict__ r0b1, const float* __restrict__ r0b2,
    const float* __restrict__ r1W1, const float* __restrict__ r1W2,
    const float* __restrict__ r1b1, const float* __restrict__ r1b2,
    const float* __restrict__ r2W0, const float* __restrict__ r2W1,
    const float* __restrict__ r2W2, const float* __restrict__ r2b0,
    const float* __restrict__ r2b1, const float* __restrict__ r2b2,
    const float* __restrict__ fW, const float* __restrict__ fb,
    float* __restrict__ out) {
  __shared__ __align__(32) _Float16 sW[12288];
  __shared__ float sB[256];
  __shared__ float sFi[40];
  __shared__ __align__(32) _Float16 sX[WPB][16 * 96];
  __shared__ __align__(32) _Float16 sH[WPB][16 * 32];
  __shared__ __align__(32) _Float16 sA[WPB][16 * 32];

  const int i = blockIdx.x;
  const int tid = threadIdx.x;

  cvtW(sW + OW1_0, r0W1, 84, 32, 96, 32, tid, 128);
  cvtW(sW + OW0_0, r0W0, 84, 32, 96, 32, tid, 128);
  cvtW(sW + OW2_0, r0W2, 32, 32, 32, 32, tid, 128);
  cvtW(sW + OW1_1, r1W1, 32, 32, 32, 32, tid, 128);
  cvtW(sW + OW2_1, r1W2, 32, 32, 32, 32, tid, 128);
  cvtW(sW + OW1_2, r2W1, 32, 16, 32, 16, tid, 128);
  cvtW(sW + OW0_2, r2W0, 32, 16, 32, 16, tid, 128);
  cvtW(sW + OW2_2, r2W2, 16, 16, 32, 16, tid, 128);
  cvtW(sW + OWF, fW, 16, 40, 32, 48, tid, 128);
  cvtB(sB + BB1_0, r0b1, 32, 32, tid, 128);
  cvtB(sB + BB0_0, r0b0, 32, 32, tid, 128);
  cvtB(sB + BB2_0, r0b2, 32, 32, tid, 128);
  cvtB(sB + BB1_1, r1b1, 32, 32, tid, 128);
  cvtB(sB + BB2_1, r1b2, 32, 32, tid, 128);
  cvtB(sB + BB1_2, r2b1, 16, 16, tid, 128);
  cvtB(sB + BB0_2, r2b0, 16, 16, tid, 128);
  cvtB(sB + BB2_2, r2b2, 16, 16, tid, 128);
  cvtB(sB + BBF, fb, 40, 48, tid, 128);
  for (int c = tid; c < 40; c += 128) sFi[c] = feat[i * 40 + c];
  __syncthreads();

  const int wave = tid >> 5, lane = tid & 31;
  const int m = lane & 15;
  const bool hi = lane >= 16;
  const int mb = hi ? 8 : 0;  // D-fragment row base
  _Float16* Xw = sX[wave];
  _Float16* Hw = sH[wave];
  _Float16* Aw = sA[wave];

  const float pix = pc[i * 3 + 0], piy = pc[i * 3 + 1], piz = pc[i * 3 + 2];
  const float nix = nrm[i * 3 + 0], niy = nrm[i * 3 + 1], niz = nrm[i * 3 + 2];

  for (int t = wave; t < NPTS / 16; t += WPB) {
    const int j0 = t * 16;
    // ---- build X (16 x 96 f16): [feat_i | feat_j | ppf4 | pad] ----
    {
      const int j = j0 + m;
      if (!hi) {
        for (int c = 0; c < 40; ++c) Xw[m * 96 + c] = (_Float16)sFi[c];
        for (int c = 40; c < 48; ++c)
          Xw[m * 96 + c] = (_Float16)feat[j * 40 + (c - 40)];
      } else {
        for (int c = 48; c < 80; ++c)
          Xw[m * 96 + c] = (_Float16)feat[j * 40 + (c - 40)];
        float d = dist[i * NPTS + j];
        float inv = 1.0f / (d + 1e-7f);
        float x0 = (pix - pc[j * 3 + 0]) * inv;
        float x1 = (piy - pc[j * 3 + 1]) * inv;
        float x2 = (piz - pc[j * 3 + 2]) * inv;
        float njx = nrm[j * 3 + 0], njy = nrm[j * 3 + 1], njz = nrm[j * 3 + 2];
        Xw[m * 96 + 80] = (_Float16)(nix * x0 + niy * x1 + niz * x2);
        Xw[m * 96 + 81] = (_Float16)(njx * x0 + njy * x1 + njz * x2);
        Xw[m * 96 + 82] = (_Float16)(nix * njx + niy * njy + niz * njz);
        Xw[m * 96 + 83] = (_Float16)d;
        for (int c = 84; c < 96; ++c) Xw[m * 96 + c] = (_Float16)0.0f;
      }
    }
    lds_fence();

    // ---- res0: h1 = relu(X@W1+b1); a1 = h1@W2+b2 + (X@W0+b0) ----
    half16 ax0 = ldA(Xw, 96, m, 0, hi);
    half16 ax1 = ldA(Xw, 96, m, 32, hi);
    half16 ax2 = ldA(Xw, 96, m, 64, hi);
    float8 acc0[2];
#pragma unroll
    for (int nt = 0; nt < 2; ++nt) {
      const int n = m + nt * 16;
      float8 c1 = f8zero(), c0 = f8zero();
      c1 = wmma16(ax0, ldB(sW + OW1_0, 96, n, 0, hi), c1);
      c1 = wmma16(ax1, ldB(sW + OW1_0, 96, n, 32, hi), c1);
      c1 = wmma16(ax2, ldB(sW + OW1_0, 96, n, 64, hi), c1);
      c0 = wmma16(ax0, ldB(sW + OW0_0, 96, n, 0, hi), c0);
      c0 = wmma16(ax1, ldB(sW + OW0_0, 96, n, 32, hi), c0);
      c0 = wmma16(ax2, ldB(sW + OW0_0, 96, n, 64, hi), c0);
      const float b1v = sB[BB1_0 + n], b0v = sB[BB0_0 + n];
#pragma unroll
      for (int r = 0; r < 8; ++r) {
        Hw[(mb + r) * 32 + n] = (_Float16)fmaxf(c1[r] + b1v, 0.f);
        c0[r] += b0v;
      }
      acc0[nt] = c0;
    }
    lds_fence();
    half16 ah = ldA(Hw, 32, m, 0, hi);
    float8 a1[2];
#pragma unroll
    for (int nt = 0; nt < 2; ++nt) {
      const int n = m + nt * 16;
      float8 c = f8zero();
      c = wmma16(ah, ldB(sW + OW2_0, 32, n, 0, hi), c);
      const float b2v = sB[BB2_0 + n];
#pragma unroll
      for (int r = 0; r < 8; ++r) {
        c[r] += b2v + acc0[nt][r];
        Aw[(mb + r) * 32 + n] = (_Float16)c[r];
      }
      a1[nt] = c;
    }
    lds_fence();

    // ---- res1: a2 = relu(a1@W1+b1)@W2+b2 + a1 ----
    half16 aa = ldA(Aw, 32, m, 0, hi);
#pragma unroll
    for (int nt = 0; nt < 2; ++nt) {
      const int n = m + nt * 16;
      float8 c = f8zero();
      c = wmma16(aa, ldB(sW + OW1_1, 32, n, 0, hi), c);
      const float bv = sB[BB1_1 + n];
#pragma unroll
      for (int r = 0; r < 8; ++r)
        Hw[(mb + r) * 32 + n] = (_Float16)fmaxf(c[r] + bv, 0.f);
    }
    lds_fence();
    ah = ldA(Hw, 32, m, 0, hi);
#pragma unroll
    for (int nt = 0; nt < 2; ++nt) {
      const int n = m + nt * 16;
      float8 c = f8zero();
      c = wmma16(ah, ldB(sW + OW2_1, 32, n, 0, hi), c);
      const float bv = sB[BB2_1 + n];
#pragma unroll
      for (int r = 0; r < 8; ++r) {
        c[r] += bv + a1[nt][r];
        Aw[(mb + r) * 32 + n] = (_Float16)c[r];
      }
    }
    lds_fence();

    // ---- res2 (32->16): a3 = relu(a2@W1+b1)@W2+b2 + (a2@W0+b0) ----
    aa = ldA(Aw, 32, m, 0, hi);
    {
      float8 chh = f8zero(), cxx = f8zero();
      chh = wmma16(aa, ldB(sW + OW1_2, 32, m, 0, hi), chh);
      cxx = wmma16(aa, ldB(sW + OW0_2, 32, m, 0, hi), cxx);
      const float b1v = sB[BB1_2 + m], b0v = sB[BB0_2 + m];
#pragma unroll
      for (int r = 0; r < 8; ++r) {
        Hw[(mb + r) * 32 + m] = (_Float16)fmaxf(chh[r] + b1v, 0.f);
        cxx[r] += b0v;
      }
      lds_fence();
      ah = ldA(Hw, 32, m, 0, hi);  // cols>=16 stale but W2_2 K-pad rows are 0
      float8 a3 = f8zero();
      a3 = wmma16(ah, ldB(sW + OW2_2, 32, m, 0, hi), a3);
      const float b2v = sB[BB2_2 + m];
#pragma unroll
      for (int r = 0; r < 8; ++r) {
        a3[r] += b2v + cxx[r];
        Aw[(mb + r) * 32 + m] = (_Float16)a3[r];
      }
    }
    lds_fence();

    // ---- final (16->40): out = a3@Wf + bf ----
    aa = ldA(Aw, 32, m, 0, hi);  // cols>=16 stale but Wf K-pad rows are 0
#pragma unroll
    for (int nt = 0; nt < 3; ++nt) {
      const int col = m + nt * 16;
      float8 o = f8zero();
      o = wmma16(aa, ldB(sW + OWF, 32, col, 0, hi), o);
      if (col < 40) {
        const float bv = sB[BBF + col];
        const size_t rowb = (size_t)i * NPTS + j0 + mb;
#pragma unroll
        for (int r = 0; r < 8; ++r) out[(rowb + r) * 40 + col] = o[r] + bv;
      }
    }
  }
}

// ---------------------------------------------------------------------------
extern "C" void kernel_launch(void* const* d_in, const int* in_sizes, int n_in,
                              void* d_out, int out_size, void* d_ws,
                              size_t ws_size, hipStream_t stream) {
  (void)in_sizes; (void)n_in; (void)out_size; (void)ws_size;
  // input order: pc, pc_normal, dist, then params flattened with sorted dict
  // keys: aggr0{W,b}, aggr1{W,b}, final{W,b},
  // res0{W0,W1,W2,b0,b1,b2}, res1{W1,W2,b1,b2}, res2{W0,W1,W2,b0,b1,b2},
  // spconv0{kW0..4, kb0..4, oW, ob}, spconv1{kW0..4, kb0..4, oW, ob}
  const float* pc = (const float*)d_in[0];
  const float* pcn = (const float*)d_in[1];
  const float* dist = (const float*)d_in[2];
  const float* a0W = (const float*)d_in[3];
  const float* a0b = (const float*)d_in[4];
  const float* a1W = (const float*)d_in[5];
  const float* a1b = (const float*)d_in[6];
  const float* fiW = (const float*)d_in[7];
  const float* fib = (const float*)d_in[8];
  const float* r0W0 = (const float*)d_in[9];
  const float* r0W1 = (const float*)d_in[10];
  const float* r0W2 = (const float*)d_in[11];
  const float* r0b0 = (const float*)d_in[12];
  const float* r0b1 = (const float*)d_in[13];
  const float* r0b2 = (const float*)d_in[14];
  const float* r1W1 = (const float*)d_in[15];
  const float* r1W2 = (const float*)d_in[16];
  const float* r1b1 = (const float*)d_in[17];
  const float* r1b2 = (const float*)d_in[18];
  const float* r2W0 = (const float*)d_in[19];
  const float* r2W1 = (const float*)d_in[20];
  const float* r2W2 = (const float*)d_in[21];
  const float* r2b0 = (const float*)d_in[22];
  const float* r2b1 = (const float*)d_in[23];
  const float* r2b2 = (const float*)d_in[24];
  const float* s0kW[5] = {(const float*)d_in[25], (const float*)d_in[26],
                          (const float*)d_in[27], (const float*)d_in[28],
                          (const float*)d_in[29]};
  const float* s0kb[5] = {(const float*)d_in[30], (const float*)d_in[31],
                          (const float*)d_in[32], (const float*)d_in[33],
                          (const float*)d_in[34]};
  const float* s0oW = (const float*)d_in[35];
  const float* s0ob = (const float*)d_in[36];
  const float* s1kW[5] = {(const float*)d_in[37], (const float*)d_in[38],
                          (const float*)d_in[39], (const float*)d_in[40],
                          (const float*)d_in[41]};
  const float* s1kb[5] = {(const float*)d_in[42], (const float*)d_in[43],
                          (const float*)d_in[44], (const float*)d_in[45],
                          (const float*)d_in[46]};
  const float* s1oW = (const float*)d_in[47];
  const float* s1ob = (const float*)d_in[48];

  char* ws = (char*)d_ws;
  size_t off = 0;
  auto alloc = [&](size_t bytes) -> void* {
    void* p = ws + off;
    off += (bytes + 255) & ~(size_t)255;
    return p;
  };
  int* nbrs = (int*)alloc(NPTS * KNN * 4);
  float* kern0 = (float*)alloc((size_t)NPTS * KNN * 32 * 4);
  float* kern1 = (float*)alloc((size_t)NPTS * KNN * 32 * 4);
  float* nf0 = (float*)alloc((size_t)NPTS * KNN * 2 * 4);
  float* nf1 = (float*)alloc((size_t)NPTS * KNN * 40 * 4);
  float* convout = (float*)alloc((size_t)NPTS * 32 * 4);
  float* g = (float*)alloc((size_t)NPTS * 8 * 4);
  float* gmax = (float*)alloc(8 * 4);
  float* feat0 = (float*)alloc((size_t)NPTS * 40 * 4);
  float* feat1 = (float*)alloc((size_t)NPTS * 40 * 4);
  float* outp = (float*)d_out;

  topk_kernel<<<NPTS, 64, 0, stream>>>(dist, nbrs);

  kernmlp_kernel<<<NPTS * KNN / 8, 256, 0, stream>>>(
      pc, nbrs, s0kW[0], s0kb[0], s0kW[1], s0kb[1], s0kW[2], s0kb[2], s0kW[3],
      s0kb[3], s0kW[4], s0kb[4], kern0);
  nf0_kernel<<<(NPTS * KNN + 255) / 256, 256, 0, stream>>>(pc, pcn, nbrs, nf0);
  conv_out_kernel<<<NPTS, 128, 0, stream>>>(kern0, nf0, 2, s0oW, s0ob, a0W,
                                            a0b, convout, g);
  gmax_kernel<<<1, 256, 0, stream>>>(g, gmax);
  feat_kernel<<<(NPTS * 40 + 255) / 256, 256, 0, stream>>>(convout, gmax,
                                                           feat0);

  kernmlp_kernel<<<NPTS * KNN / 8, 256, 0, stream>>>(
      pc, nbrs, s1kW[0], s1kb[0], s1kW[1], s1kb[1], s1kW[2], s1kb[2], s1kW[3],
      s1kb[3], s1kW[4], s1kb[4], kern1);
  gather_feat_kernel<<<(NPTS * KNN * 40 + 255) / 256, 256, 0, stream>>>(
      feat0, nbrs, nf1);
  conv_out_kernel<<<NPTS, 128, 0, stream>>>(kern1, nf1, 40, s1oW, s1ob, a1W,
                                            a1b, convout, g);
  gmax_kernel<<<1, 256, 0, stream>>>(g, gmax);
  feat_kernel<<<(NPTS * 40 + 255) / 256, 256, 0, stream>>>(convout, gmax,
                                                           feat1);

  pair_mlp_kernel<<<NPTS, 128, 0, stream>>>(
      pc, pcn, dist, feat1, r0W0, r0W1, r0W2, r0b0, r0b1, r0b2, r1W1, r1W2,
      r1b1, r1b2, r2W0, r2W1, r2W2, r2b0, r2b1, r2b2, fiW, fib, outp);
}
